// HAP_44452911514227
// MI455X (gfx1250) — matrix operand order repaired and checked by reading, not compiled, not run on hardware
//
#include <hip/hip_runtime.h>
#include <hip/hip_bf16.h>

typedef __attribute__((ext_vector_type(16))) __bf16 v16bf;
typedef __attribute__((ext_vector_type(8)))  float  v8f;

#define BDIM 16
#define VDIM 48
#define SDIM 256
#define EDIM 256
#define ADIM 64
#define HDIM 256
#define ODIM 90
#define GDIM 768  // 3*H
#define VOCAB 3350
#define PAD_IDX 756  // embedding row is all zeros in the reference

// ---------- helpers ----------

static __device__ __forceinline__ unsigned short f2bf(float x) {
  __hip_bfloat16 h = __float2bfloat16(x);
  return __builtin_bit_cast(unsigned short, h);
}

union BFrag { uint4 u[2]; v16bf v; };

// A fragment: 16x32 bf16 tile, lane (l&15) = row M; K chunks {kb..kb+7, kb+16..kb+23}, kb = 0 (lanes<16) / 8 (lanes>=16)
static __device__ __forceinline__ v16bf load_a_frag(const unsigned short* rowbase, int kstep, int lane) {
  BFrag f;
  const int kb = kstep * 32 + ((lane < 16) ? 0 : 8);
  f.u[0] = *reinterpret_cast<const uint4*>(rowbase + kb);
  f.u[1] = *reinterpret_cast<const uint4*>(rowbase + kb + 16);
  return f.v;
}

// B fragment: 32x16 bf16 tile, lane (l&15) = col N; K chunks {kb..kb+7, kb+16..kb+23}, kb = 0 (lanes<16) / 8 (lanes>=16)
// colbase points at row n of a row-major (N x K) weight matrix (i.e. column n of B = W^T)
static __device__ __forceinline__ v16bf load_b_frag(const unsigned short* colbase, int kstep, int lane) {
  BFrag f;
  const int kb = kstep * 32 + ((lane < 16) ? 0 : 8);
  f.u[0] = *reinterpret_cast<const uint4*>(colbase + kb);
  f.u[1] = *reinterpret_cast<const uint4*>(colbase + kb + 16);
  return f.v;
}

static __device__ __forceinline__ float sigmoidf_(float x) { return 1.0f / (1.0f + expf(-x)); }

// ---------- kernel 0: fp32 -> bf16 weight/table conversion ----------

__global__ __launch_bounds__(256) void k_convert(const float* __restrict__ emb, const float* __restrict__ wa,
                                                 const float* __restrict__ wih, const float* __restrict__ whh,
                                                 unsigned short* __restrict__ oe, unsigned short* __restrict__ owa,
                                                 unsigned short* __restrict__ owi, unsigned short* __restrict__ owh) {
  const int i = blockIdx.x * blockDim.x + threadIdx.x;
  const int stride = gridDim.x * blockDim.x;
  for (int j = i; j < VOCAB * EDIM; j += stride) oe[j] = f2bf(emb[j]);
  for (int j = i; j < ADIM * 2 * EDIM; j += stride) owa[j] = f2bf(wa[j]);
  for (int j = i; j < GDIM * EDIM; j += stride) owi[j] = f2bf(wih[j]);
  for (int j = i; j < GDIM * HDIM; j += stride) owh[j] = f2bf(whh[j]);
}

// ---------- kernel 1: dual attention + softmax + weighted embedding, one block per (b,v) ----------

__global__ __launch_bounds__(256) void k_attn(const int* __restrict__ x, const int* __restrict__ p2c,
                                              const int* __restrict__ c2p, const float* __restrict__ indx,
                                              const float* __restrict__ ind1, const float* __restrict__ ind2,
                                              const float* __restrict__ emb_f32, const float* __restrict__ wa_b,
                                              const float* __restrict__ ua, const unsigned short* __restrict__ emb_bf,
                                              const unsigned short* __restrict__ wa_bf,
                                              unsigned short* __restrict__ embw_bf) {
  __shared__ unsigned short s_wa[ADIM * 2 * EDIM];  // 64 KB: Wa_w bf16, row-major (64 x 512)
  __shared__ int s_tx[SDIM], s_tp[SDIM], s_tc[SDIM];
  __shared__ float s_ix[SDIM];
  __shared__ float s_sc1[SDIM], s_sc2[SDIM];
  __shared__ float s_red[SDIM];
  __shared__ float s_w[SDIM];

  const int bv = blockIdx.x;
  const int tid = threadIdx.x;
  const int lane = tid & 31;
  const int wv = tid >> 5;

  // cooperative stage of Wa_w (bf16) into LDS: 4096 uint4, 16 per thread
  {
    const uint4* src = reinterpret_cast<const uint4*>(wa_bf);
    uint4* dst = reinterpret_cast<uint4*>(s_wa);
#pragma unroll 1
    for (int j = tid; j < (ADIM * 2 * EDIM) / 8; j += 256) dst[j] = src[j];
  }
  {
    const int g = bv * SDIM + tid;
    const float iX = indx[g], i1 = ind1[g], i2 = ind2[g];
    // indicator==0 -> redirect to PAD row (all-zero embedding): branch-free masking
    s_tx[tid] = (iX == 0.0f) ? PAD_IDX : x[g];
    s_tp[tid] = (i1 == 0.0f) ? PAD_IDX : p2c[g];
    s_tc[tid] = (i2 == 0.0f) ? PAD_IDX : c2p[g];
    s_ix[tid] = iX;
  }
  __syncthreads();

  float wab_l[4], ua_l[4];
#pragma unroll
  for (int nt = 0; nt < 4; ++nt) {
    const int n = nt * 16 + (lane & 15);
    wab_l[nt] = wa_b[n];
    ua_l[nt] = ua[n];
  }

  // each wave owns 2 token M-tiles (16 tiles total)
#pragma unroll 1
  for (int tt = 0; tt < 2; ++tt) {
    const int tile = wv * 2 + tt;
    const int trow = tile * 16 + (lane & 15);
    const unsigned short* ax_base = emb_bf + (size_t)s_tx[trow] * EDIM;
    const unsigned short* ap_base = emb_bf + (size_t)s_tp[trow] * EDIM;
    const unsigned short* ac_base = emb_bf + (size_t)s_tc[trow] * EDIM;

    // ---- phase 1: Cx = e_x @ Wx^T ; Cp = e_p2c @ Wp^T  (K = 256 -> 8 steps of 32) ----
    v8f Cx[4] = {}, Cp[4] = {};
#pragma unroll 1
    for (int k = 0; k < 8; ++k) {
      const v16bf ax = load_a_frag(ax_base, k, lane);
      const v16bf ap = load_a_frag(ap_base, k, lane);
      v16bf bx[4];
#pragma unroll
      for (int nt = 0; nt < 4; ++nt) bx[nt] = load_b_frag(s_wa + (nt * 16 + (lane & 15)) * 512, k, lane);
#pragma unroll
      for (int nt = 0; nt < 4; ++nt)
        Cx[nt] = __builtin_amdgcn_wmma_f32_16x16x32_bf16(false, ax, false, bx[nt], (short)0, Cx[nt], false, false);
      v16bf bp[4];
#pragma unroll
      for (int nt = 0; nt < 4; ++nt) bp[nt] = load_b_frag(s_wa + (nt * 16 + (lane & 15)) * 512 + 256, k, lane);
#pragma unroll
      for (int nt = 0; nt < 4; ++nt)
        Cp[nt] = __builtin_amdgcn_wmma_f32_16x16x32_bf16(false, ap, false, bp[nt], (short)0, Cp[nt], false, false);
    }

    // score_p2c[M] = sum_n ua[n] * tanh(Cx + Cp + wa_b[n])
    float part[8];
#pragma unroll
    for (int r = 0; r < 8; ++r) part[r] = 0.0f;
#pragma unroll
    for (int nt = 0; nt < 4; ++nt) {
#pragma unroll
      for (int r = 0; r < 8; ++r) {
        const float t = tanhf(Cx[nt][r] + Cp[nt][r] + wab_l[nt]);
        part[r] += t * ua_l[nt];
      }
    }
#pragma unroll
    for (int r = 0; r < 8; ++r) {
#pragma unroll
      for (int m = 8; m >= 1; m >>= 1) part[r] += __shfl_xor(part[r], m, 32);
    }
    if ((lane & 15) == 0) {
      const int mb = tile * 16 + ((lane >= 16) ? 8 : 0);
#pragma unroll
      for (int r = 0; r < 8; ++r) s_sc1[mb + r] = part[r];
    }

    // ---- phase 2: Cc = e_c2p @ Wp^T, reuse Cx ----
    v8f Cc[4] = {};
#pragma unroll 1
    for (int k = 0; k < 8; ++k) {
      const v16bf ac = load_a_frag(ac_base, k, lane);
      v16bf bp[4];
#pragma unroll
      for (int nt = 0; nt < 4; ++nt) bp[nt] = load_b_frag(s_wa + (nt * 16 + (lane & 15)) * 512 + 256, k, lane);
#pragma unroll
      for (int nt = 0; nt < 4; ++nt)
        Cc[nt] = __builtin_amdgcn_wmma_f32_16x16x32_bf16(false, ac, false, bp[nt], (short)0, Cc[nt], false, false);
    }
#pragma unroll
    for (int r = 0; r < 8; ++r) part[r] = 0.0f;
#pragma unroll
    for (int nt = 0; nt < 4; ++nt) {
#pragma unroll
      for (int r = 0; r < 8; ++r) {
        const float t = tanhf(Cx[nt][r] + Cc[nt][r] + wab_l[nt]);
        part[r] += t * ua_l[nt];
      }
    }
#pragma unroll
    for (int r = 0; r < 8; ++r) {
#pragma unroll
      for (int m = 8; m >= 1; m >>= 1) part[r] += __shfl_xor(part[r], m, 32);
    }
    if ((lane & 15) == 0) {
      const int mb = tile * 16 + ((lane >= 16) ? 8 : 0);
#pragma unroll
      for (int r = 0; r < 8; ++r) s_sc2[mb + r] = part[r];
    }
  }
  __syncthreads();

  // ---- softmax over S for both score sets ----
  const float v1 = s_sc1[tid], v2 = s_sc2[tid];
  s_red[tid] = v1;
  __syncthreads();
  for (int off = 128; off > 0; off >>= 1) {
    if (tid < off) s_red[tid] = fmaxf(s_red[tid], s_red[tid + off]);
    __syncthreads();
  }
  const float m1 = s_red[0];
  __syncthreads();
  const float e1 = expf(v1 - m1);
  s_red[tid] = e1;
  __syncthreads();
  for (int off = 128; off > 0; off >>= 1) {
    if (tid < off) s_red[tid] += s_red[tid + off];
    __syncthreads();
  }
  const float sm1 = s_red[0];
  __syncthreads();
  s_red[tid] = v2;
  __syncthreads();
  for (int off = 128; off > 0; off >>= 1) {
    if (tid < off) s_red[tid] = fmaxf(s_red[tid], s_red[tid + off]);
    __syncthreads();
  }
  const float m2 = s_red[0];
  __syncthreads();
  const float e2 = expf(v2 - m2);
  s_red[tid] = e2;
  __syncthreads();
  for (int off = 128; off > 0; off >>= 1) {
    if (tid < off) s_red[tid] += s_red[tid + off];
    __syncthreads();
  }
  const float sm2 = s_red[0];
  __syncthreads();

  // combined weight; indicator_x already folded in via PAD redirection of s_tx
  // (values are attended over emb_x in both attentions; zero row contributes zero)
  s_w[tid] = (e1 / sm1 + e2 / sm2) * s_ix[tid];
  __syncthreads();

  // ---- weighted sum over tokens (fp32 table for accuracy); thread = E column ----
  float acc = 0.0f;
#pragma unroll 4
  for (int l = 0; l < SDIM; ++l) {
    acc += s_w[l] * emb_f32[(size_t)s_tx[l] * EDIM + tid];
  }
  embw_bf[(size_t)bv * EDIM + tid] = f2bf(acc);
}

// ---------- kernel 2: x_proj = emb_weighted @ W_ih^T + b_ih  (768 x 768, K=256) ----------

__global__ __launch_bounds__(256) void k_xproj(const unsigned short* __restrict__ embw_bf,
                                               const unsigned short* __restrict__ wih_bf,
                                               const float* __restrict__ b_ih, float* __restrict__ xproj) {
  const int tid = threadIdx.x, lane = tid & 31, wv = tid >> 5;
  const int mtile = blockIdx.x;  // 48 M-tiles
  const unsigned short* abase = embw_bf + (size_t)(mtile * 16 + (lane & 15)) * EDIM;

  v8f C[6] = {};
#pragma unroll 1
  for (int k = 0; k < 8; ++k) {
    const v16bf a = load_a_frag(abase, k, lane);
    v16bf bf[6];
#pragma unroll
    for (int nt = 0; nt < 6; ++nt)
      bf[nt] = load_b_frag(wih_bf + (size_t)((wv * 6 + nt) * 16 + (lane & 15)) * EDIM, k, lane);
#pragma unroll
    for (int nt = 0; nt < 6; ++nt)
      C[nt] = __builtin_amdgcn_wmma_f32_16x16x32_bf16(false, a, false, bf[nt], (short)0, C[nt], false, false);
  }
#pragma unroll
  for (int nt = 0; nt < 6; ++nt) {
    const int col = (wv * 6 + nt) * 16 + (lane & 15);
    const float bb = b_ih[col];
#pragma unroll
    for (int r = 0; r < 8; ++r) {
      const int row = mtile * 16 + r + ((lane >= 16) ? 8 : 0);
      xproj[(size_t)row * GDIM + col] = C[nt][r] + bb;
    }
  }
}

// ---------- kernel 3: GRU scan over V (single block) + head ----------

__global__ __launch_bounds__(256) void k_gru(const float* __restrict__ xproj, const unsigned short* __restrict__ whh_bf,
                                             const float* __restrict__ b_hh, const float* __restrict__ lin_w,
                                             const float* __restrict__ lin_b, float* __restrict__ out) {
  __shared__ unsigned short s_hbf[BDIM * HDIM];  // 8 KB  (bf16 h for WMMA A)
  __shared__ float s_gh[BDIM * GDIM];            // 48 KB (h @ W_hh^T)

  const int tid = threadIdx.x, lane = tid & 31, wv = tid >> 5;
  const int gb = tid >> 4;   // batch row owned in gate phase
  const int gj = tid & 15;   // hd = gj + 16*i

  float h_reg[16], rs_reg[16];
#pragma unroll
  for (int i = 0; i < 16; ++i) {
    h_reg[i] = 0.0f;
    rs_reg[i] = 0.0f;
  }
  for (int i = tid; i < BDIM * HDIM; i += 256) s_hbf[i] = 0;
  __syncthreads();

  const unsigned short* abase = s_hbf + (lane & 15) * HDIM;  // A row = batch

#pragma unroll 1
  for (int v = 0; v < VDIM; ++v) {
    // gh = h @ W_hh^T : single M-tile (batch=16), 48 N-tiles split across 8 waves
    v8f C[6] = {};
#pragma unroll 1
    for (int k = 0; k < 8; ++k) {
      const v16bf a = load_a_frag(abase, k, lane);
      v16bf bf[6];
#pragma unroll
      for (int nt = 0; nt < 6; ++nt)
        bf[nt] = load_b_frag(whh_bf + (size_t)((wv * 6 + nt) * 16 + (lane & 15)) * HDIM, k, lane);
#pragma unroll
      for (int nt = 0; nt < 6; ++nt)
        C[nt] = __builtin_amdgcn_wmma_f32_16x16x32_bf16(false, a, false, bf[nt], (short)0, C[nt], false, false);
    }
#pragma unroll
    for (int nt = 0; nt < 6; ++nt) {
      const int col = (wv * 6 + nt) * 16 + (lane & 15);
#pragma unroll
      for (int r = 0; r < 8; ++r) {
        const int row = r + ((lane >= 16) ? 8 : 0);
        s_gh[row * GDIM + col] = C[nt][r];
      }
    }
    __syncthreads();

    // gate math: thread owns (gb, gj+16i), i=0..15
    const float* xp = xproj + (size_t)(gb * VDIM + v) * GDIM;
#pragma unroll
    for (int i = 0; i < 16; ++i) {
      const int hd = gj + 16 * i;
      const float hr = s_gh[gb * GDIM + hd] + b_hh[hd];
      const float hz = s_gh[gb * GDIM + HDIM + hd] + b_hh[HDIM + hd];
      const float hn = s_gh[gb * GDIM + 2 * HDIM + hd] + b_hh[2 * HDIM + hd];
      const float r_ = sigmoidf_(xp[hd] + hr);
      const float z_ = sigmoidf_(xp[HDIM + hd] + hz);
      const float n_ = tanhf(xp[2 * HDIM + hd] + r_ * hn);
      const float hnew = (1.0f - z_) * n_ + z_ * h_reg[i];
      h_reg[i] = hnew;
      rs_reg[i] += hnew;
      s_hbf[gb * HDIM + hd] = f2bf(hnew);
    }
    __syncthreads();
  }

  // head: out[b,o] = sigmoid( dot(rsum[b], lin_w[o]) + V*lin_b[o] )
  float* s_rs = s_gh;  // reuse
#pragma unroll
  for (int i = 0; i < 16; ++i) s_rs[gb * HDIM + gj + 16 * i] = rs_reg[i];
  __syncthreads();

#pragma unroll 1
  for (int idx = tid; idx < BDIM * ODIM; idx += 256) {
    const int b = idx / ODIM, o = idx % ODIM;
    float acc = (float)VDIM * lin_b[o];
#pragma unroll 4
    for (int k = 0; k < HDIM; ++k) acc += s_rs[b * HDIM + k] * lin_w[o * HDIM + k];
    out[idx] = sigmoidf_(acc);
  }
}

// ---------- host ----------

extern "C" void kernel_launch(void* const* d_in, const int* in_sizes, int n_in, void* d_out, int out_size, void* d_ws,
                              size_t ws_size, hipStream_t stream) {
  (void)in_sizes; (void)n_in; (void)out_size; (void)ws_size;
  const int* x = (const int*)d_in[0];
  const int* p2c = (const int*)d_in[1];
  const int* c2p = (const int*)d_in[2];
  const float* indx = (const float*)d_in[3];
  const float* ind1 = (const float*)d_in[4];
  const float* ind2 = (const float*)d_in[5];
  const float* emb = (const float*)d_in[6];
  const float* wa_w = (const float*)d_in[7];
  const float* wa_b = (const float*)d_in[8];
  const float* ua = (const float*)d_in[9];
  const float* w_ih = (const float*)d_in[10];
  const float* w_hh = (const float*)d_in[11];
  const float* b_ih = (const float*)d_in[12];
  const float* b_hh = (const float*)d_in[13];
  const float* lin_w = (const float*)d_in[14];
  const float* lin_b = (const float*)d_in[15];
  float* out = (float*)d_out;

  char* ws = (char*)d_ws;
  // workspace layout (256B aligned segments)
  unsigned short* emb_bf = (unsigned short*)(ws + 0);              // 3350*256*2  = 1,715,200
  unsigned short* wa_bf = (unsigned short*)(ws + 1715200);         // 64*512*2    =    65,536
  unsigned short* wih_bf = (unsigned short*)(ws + 1780736);        // 768*256*2   =   393,216
  unsigned short* whh_bf = (unsigned short*)(ws + 2173952);        // 768*256*2   =   393,216
  unsigned short* embw_bf = (unsigned short*)(ws + 2567168);       // 768*256*2   =   393,216
  float* xproj = (float*)(ws + 2960384);                           // 768*768*4   = 2,359,296 -> total ~5.32 MB

  k_convert<<<dim3((VOCAB * EDIM + 255) / 256), dim3(256), 0, stream>>>(emb, wa_w, w_ih, w_hh, emb_bf, wa_bf, wih_bf,
                                                                        whh_bf);
  k_attn<<<dim3(BDIM * VDIM), dim3(256), 0, stream>>>(x, p2c, c2p, indx, ind1, ind2, emb, wa_b, ua, emb_bf, wa_bf,
                                                      embw_bf);
  k_xproj<<<dim3(48), dim3(256), 0, stream>>>(embw_bf, wih_bf, b_ih, xproj);
  k_gru<<<dim3(1), dim3(256), 0, stream>>>(xproj, whh_bf, b_hh, lin_w, lin_b, out);
}
